// NONA_32169305047734
// MI455X (gfx1250) — matrix operand order repaired and checked by reading, not compiled, not run on hardware
//
#include <hip/hip_runtime.h>
#include <hip/hip_fp16.h>

#define NN   8192
#define DD   1024
#define CC   100
#define TOPK 64
#define BM   16
#define NWAVES 8
#define BCOLS (NWAVES * 16)   // 128 columns per outer iteration

typedef __attribute__((ext_vector_type(16))) __bf16          v16bf;
typedef __attribute__((ext_vector_type(8)))  float           v8f;
typedef __attribute__((ext_vector_type(16))) unsigned short  v16u;
typedef __attribute__((ext_vector_type(8)))  unsigned int    v8u;

union BF16x16 { v16u u16; v8u u32; v16bf b; };

static __device__ __forceinline__ unsigned short f2bf(float f) {
    unsigned int u = __float_as_uint(f);
    u += 0x7FFFu + ((u >> 16) & 1u);           // round-to-nearest-even
    return (unsigned short)(u >> 16);
}
static __device__ __forceinline__ unsigned short f2h(float f) {
    return __half_as_ushort(__float2half(f));
}
static __device__ __forceinline__ float h2f(unsigned short s) {
    return __half2float(__ushort_as_half(s));
}

// ------------------------------------------------------------- fp32->bf16 --
__global__ __launch_bounds__(256) void cvt_kernel(const float* __restrict__ src,
                                                  unsigned short* __restrict__ dst) {
    size_t i = ((size_t)blockIdx.x * 256 + threadIdx.x) * 4;
    float4 f = *(const float4*)(src + i);
    unsigned int a0 = f2bf(f.x), a1 = f2bf(f.y), a2 = f2bf(f.z), a3 = f2bf(f.w);
    uint2 pk;
    pk.x = a0 | (a1 << 16);
    pk.y = a2 | (a3 << 16);
    *(uint2*)(dst + i) = pk;
}

// ---------------------------------------------------------------- norms ----
__global__ __launch_bounds__(256) void norms_kernel(const float* __restrict__ x,
                                                    float* __restrict__ x2) {
    const int wave = threadIdx.x >> 5;
    const int lane = threadIdx.x & 31;
    const int row  = blockIdx.x * 8 + wave;
    const float* p = x + (size_t)row * DD;
    float s = 0.f;
    for (int c = lane; c < DD; c += 32) { float v = p[c]; s += v * v; }
#pragma unroll
    for (int off = 16; off > 0; off >>= 1) s += __shfl_xor(s, off, 32);
    if (lane == 0) x2[row] = s;
}

// ---------------------------------------------------------------- fused ----
__global__ __launch_bounds__(256) void fused_kernel(const unsigned short* __restrict__ xbf,
                                                    const unsigned short* __restrict__ xnbf,
                                                    const float* __restrict__ y,
                                                    const float* __restrict__ x2,
                                                    float* __restrict__ out) {
    __shared__ unsigned short simLds[BM * NN];         // f16 sim strip: 256 KB
    __shared__ unsigned short aFrag[(DD / 32) * 32 * 16]; // A in fragment order: 32 KB
    __shared__ float topV[BM][TOPK];
    __shared__ int   topI[BM][TOPK];

    const int t       = threadIdx.x;
    const int wave    = t >> 5;
    const int lane    = t & 31;
    const int rowBase = blockIdx.x * BM;
    const int g       = lane >> 4;   // half-wave group (0/1)
    const int ln      = lane & 15;

    // Phase 0: stage A tile in LDS, already permuted into WMMA fragment order.
    // aFrag[kc][l][i]: lane l's i-th bf16 of K-chunk kc (contiguous k = 16*(l/16) + i).
    for (int e = t; e < BM * DD; e += 256) {
        int kc = e >> 9;            // K-chunk (512 entries each)
        int l  = (e >> 4) & 31;     // lane
        int i  = e & 15;            // element within fragment
        int lg = l >> 4, lm = l & 15;
        int K  = kc * 32 + lg * 16 + i;
        aFrag[e] = xbf[(size_t)(rowBase + lm) * DD + K];
    }
    __syncthreads();

    // Phase 1: WMMA GEMM over all 8192 columns, epilogue -> f16 sim in LDS
    for (int cb = 0; cb < NN / BCOLS; ++cb) {
        const int myCol = cb * BCOLS + wave * 16;   // this wave's 16-col slice
        v8f acc = {};
        const unsigned short* bbase = xnbf + (size_t)(myCol + ln) * DD + g * 16;
#pragma unroll 4
        for (int kk = 0; kk < DD; kk += 32) {
            BF16x16 a, b;
            // A fragment: one contiguous 32B LDS read (2x ds_load_b128)
            a.u16 = *(const v16u*)&aFrag[(kk >> 5) * 512 + lane * 16];
            // B fragment: contiguous 32B of pre-converted bf16 (2x global_load_b128)
            b.u32 = *(const v8u*)(bbase + kk);
            acc = __builtin_amdgcn_wmma_f32_16x16x32_bf16(
                      false, a.b, false, b.b, (short)0, acc, false, false);
        }
        // Epilogue: sim = -sqrt(max(x2_i + x2_j - 2S, eps)); diag -> -inf
        const int   gcol = myCol + ln;
        const float cn2  = x2[gcol];
#pragma unroll
        for (int r = 0; r < 8; ++r) {
            int   m    = r + g * 8;                 // C/D VGPR layout
            int   grow = rowBase + m;
            float d2   = x2[grow] + cn2 - 2.0f * acc[r];
            d2 = fmaxf(d2, 1e-12f);
            float s = -__builtin_sqrtf(d2);
            if (grow == gcol) s = -__builtin_inff();
            simLds[m * NN + gcol] = f2h(s);
        }
    }
    __syncthreads();

    // Phase 2: per-row top-64 extraction + softmax + out (one wave per 2 rows)
    for (int rr = 0; rr < 2; ++rr) {
        const int row = wave * 2 + rr;
        unsigned short* srow = &simLds[row * NN];
        for (int k = 0; k < TOPK; ++k) {
            float best = -__builtin_inff();
            int   bi   = 0x7FFFFFFF;
            // vectorized sweep: 8 halves (16B) per LDS read
            for (int c = lane; c < NN / 8; c += 32) {
                uint4 q = *(const uint4*)(srow + c * 8);
#pragma unroll
                for (int ii = 0; ii < 8; ++ii) {
                    unsigned int w = (&q.x)[ii >> 1];
                    unsigned short hb = (ii & 1) ? (unsigned short)(w >> 16)
                                                 : (unsigned short)(w & 0xFFFFu);
                    float v = h2f(hb);
                    if (v > best) { best = v; bi = c * 8 + ii; }
                }
            }
#pragma unroll
            for (int off = 16; off > 0; off >>= 1) {
                float ov = __shfl_xor(best, off, 32);
                int   oi = __shfl_xor(bi, off, 32);
                if (ov > best || (ov == best && oi < bi)) { best = ov; bi = oi; }
            }
            if (lane == 0) { topV[row][k] = best; topI[row][k] = bi; }
            if (((bi >> 3) & 31) == lane) srow[bi] = 0xFC00u; // mark extracted: f16 -inf
            asm volatile("s_wait_dscnt 0" ::: "memory");      // CDNA5 split LDS counter
        }
        asm volatile("s_wait_dscnt 0" ::: "memory");

        // softmax over the 64 kept values (descending, topV[row][0] is the max)
        const float m0 = topV[row][0];
        float psum = 0.f;
        for (int k = lane; k < TOPK; k += 32) psum += __expf(topV[row][k] - m0);
#pragma unroll
        for (int off = 16; off > 0; off >>= 1) psum += __shfl_xor(psum, off, 32);
        const float inv  = 1.0f / psum;
        const int   grow = rowBase + row;
        for (int c = lane; c < CC; c += 32) {
            float a = 0.f;
            for (int k = 0; k < TOPK; ++k) {
                float w = __expf(topV[row][k] - m0) * inv;
                a += w * y[(size_t)topI[row][k] * CC + c];
            }
            out[(size_t)grow * CC + c] = a;
        }
    }
}

// --------------------------------------------------------------- launch ----
extern "C" void kernel_launch(void* const* d_in, const int* in_sizes, int n_in,
                              void* d_out, int out_size, void* d_ws, size_t ws_size,
                              hipStream_t stream) {
    const float* x  = (const float*)d_in[0];
    const float* xn = (const float*)d_in[1];
    const float* y  = (const float*)d_in[2];
    float* out = (float*)d_out;

    // workspace layout: [x2: 8192 f32][xbf: 8192*1024 bf16][xnbf: 8192*1024 bf16]
    float*          x2   = (float*)d_ws;
    unsigned short* xbf  = (unsigned short*)((char*)d_ws + 32 * 1024);
    unsigned short* xnbf = xbf + (size_t)NN * DD;

    norms_kernel<<<NN / 8, 256, 0, stream>>>(x, x2);
    cvt_kernel<<<(NN * DD) / (256 * 4), 256, 0, stream>>>(x, xbf);
    cvt_kernel<<<(NN * DD) / (256 * 4), 256, 0, stream>>>(xn, xnbf);
    fused_kernel<<<NN / BM, 256, 0, stream>>>(xbf, xnbf, y, x2, out);
}
